// SpatialWindowAttention_15899968930104
// MI455X (gfx1250) — matrix operand order repaired and checked by reading, not compiled, not run on hardware
//
#include <hip/hip_runtime.h>
#include <hip/hip_bf16.h>

// ---------------------------------------------------------------------------
// Swin shifted-window attention for MI455X (gfx1250, wave32, WMMA).
// All GEMMs on v_wmma_f32_16x16x32_bf16 (f32 accumulate). Every WMMA fragment
// is loaded from LDS as contiguous 128-bit chunks (ds_load_b128), by packing
// weights K-step-major and storing V transposed. Shifts / window partition are
// folded into load/store index math. Weights (1.2 MB bf16) stay L2-resident.
// ---------------------------------------------------------------------------

typedef __bf16 bf16;
typedef __attribute__((ext_vector_type(16))) __bf16 v16bf;
typedef __attribute__((ext_vector_type(4)))  __bf16 v4bf;
typedef __attribute__((ext_vector_type(8)))  float  v8f;

#define DIM   384
#define NHEAD 12
#define HD    32
#define WSZ   8
#define NTOK  64          // tokens per window (8*8)
#define IMG   64          // H == W
#define NWIN  2048        // 32 * 8 * 8 windows
#define SHIFT 4
#define WMAT  (DIM * DIM) // 147456
#define KSTEP 32
#define NKS   (DIM / KSTEP)   // 12 K-steps

__device__ __forceinline__ int region_id(int c) {
  // shift-mask region on the shifted grid: [0,56) / [56,60) / [60,64)
  return (c < IMG - WSZ) ? 0 : ((c < IMG - SHIFT) ? 1 : 2);
}

// ---------------------------------------------------------------------------
// CDNA5 WMMA bf16 fragment layouts (wave32), per ISA 7.12.2:
// A (16x32, MxK): lane: M=lane&15, hf=lane>>4; elements are two contiguous
//   8-runs:  K = hf*8 + e (e<8)  and  K = 16 + hf*8 + (e-8).
// B (32x16, KxN): lane: N=lane&15, hf=lane>>4; elements are two contiguous
//   8-runs:  K = hf*8 + e (e<8)  and  K = 16 + hf*8 + (e-8).   (=> store B K-contiguous per column)
// C/D (16x16 f32): lane: N=lane&15; vgpr r: M = r + (lane>=16 ? 8 : 0).
// ---------------------------------------------------------------------------
__device__ __forceinline__ v16bf load_a_frag(const bf16* row, int hf) {
  // row points at a 32-element K-contiguous row; two 16B chunks.
  v16bf a;
  ((uint4*)&a)[0] = *(const uint4*)(row + hf * 8);
  ((uint4*)&a)[1] = *(const uint4*)(row + 16 + hf * 8);
  return a;
}
__device__ __forceinline__ v16bf load_b_frag(const bf16* col, int hf) {
  // col points at a 32-element K-contiguous column; two 16B chunks.
  v16bf b;
  ((uint4*)&b)[0] = *(const uint4*)(col + hf * 8);
  ((uint4*)&b)[1] = *(const uint4*)(col + 16 + hf * 8);
  return b;
}

// ---------------------------------------------------------------------------
// Kernel 0: pack the 4 weight matrices (row-major [out,in] f32) to bf16 in
// K-step-major fragment-friendly layout: wt[p][ks][n][kk] = w[n][ks*32+kk].
// Per GEMM K-step the B stage is then a contiguous 24KB memcpy AND each
// lane's fragment is two contiguous 16B runs.
// ---------------------------------------------------------------------------
__global__ void k_pack_weights(const float* __restrict__ wq,
                               const float* __restrict__ wk,
                               const float* __restrict__ wv,
                               const float* __restrict__ wo,
                               bf16* __restrict__ wt) {
  int idx = blockIdx.x * blockDim.x + threadIdx.x;
  if (idx >= 4 * WMAT) return;
  int p   = idx / WMAT;
  int rem = idx - p * WMAT;
  int ks  = rem / (DIM * KSTEP);
  int r2  = rem - ks * (DIM * KSTEP);
  int n   = r2 >> 5;
  int kk  = r2 & 31;
  const float* w = (p == 0) ? wq : (p == 1) ? wk : (p == 2) ? wv : wo;
  wt[idx] = (bf16)w[n * DIM + ks * KSTEP + kk];
}

// ---------------------------------------------------------------------------
// Kernel 1: fused Q/K/V projection GEMM, one block per (window, proj).
// C[64 tok x 384] = A[64 x 384] * W^T, K-tiled by 32.
// 8 waves as 2(M) x 4(N): each wave owns 2 M-tiles x 6 N-tiles.
// Epilogue: +bias (Q also * hd^-0.5); Q/K go out [win][head][tok][hd],
// V goes out TRANSPOSED [win][head][hd][tok] for contiguous P@V fragments.
// ---------------------------------------------------------------------------
__global__ void __launch_bounds__(256)
k_qkv(const float* __restrict__ x, const bf16* __restrict__ wt,
      const float* __restrict__ bq, const float* __restrict__ bk,
      const float* __restrict__ bv,
      bf16* __restrict__ q_ws, bf16* __restrict__ k_ws, bf16* __restrict__ v_ws) {
  __shared__ __align__(16) bf16 As[64][KSTEP];
  __shared__ __align__(16) bf16 Bs[DIM][KSTEP];   // K-contiguous per column

  const int win = blockIdx.x;
  const int p   = blockIdx.y;              // 0=q 1=k 2=v
  const int b   = win >> 6;
  const int wy  = (win & 63) >> 3;
  const int wx  = win & 7;
  const int tid = threadIdx.x;
  const int wave = tid >> 5, lane = tid & 31;
  const int hf = lane >> 4, lm = lane & 15;
  const int waveM = wave >> 2;             // 0..1
  const int waveN = wave & 3;              // 0..3

  const bf16*  wtp  = wt + (size_t)p * WMAT;
  const float* bias = (p == 0) ? bq : (p == 1) ? bk : bv;
  bf16*        outp = (p == 0) ? q_ws : (p == 1) ? k_ws : v_ws;

  v8f zero = {};
  v8f acc[2][6];
#pragma unroll
  for (int i = 0; i < 2; ++i)
#pragma unroll
    for (int j = 0; j < 6; ++j) acc[i][j] = zero;

  for (int ks = 0; ks < NKS; ++ks) {
    const int k0 = ks * KSTEP;
    // --- A tile: 64x32, float4 loads -> packed bf16x4 stores;
    //     roll(-4,-4) folded into the gather ---
    for (int idx = tid; idx < 512; idx += 256) {
      int m = idx >> 3, q = idx & 7;
      int hs = wy * WSZ + (m >> 3), vs = wx * WSZ + (m & 7);
      int hsrc = (hs + SHIFT) & (IMG - 1);
      int wsrc = (vs + SHIFT) & (IMG - 1);
      const float4 f = ((const float4*)(x +
          (((size_t)b * IMG + hsrc) * IMG + wsrc) * DIM + k0))[q];
      v4bf h4;
      h4[0] = (bf16)f.x; h4[1] = (bf16)f.y; h4[2] = (bf16)f.z; h4[3] = (bf16)f.w;
      *(v4bf*)&As[m][q * 4] = h4;
    }
    // --- B tile: contiguous 24KB bf16 memcpy (K-step-major packed weights) ---
    {
      const uint4* src = (const uint4*)(wtp + (size_t)ks * (DIM * KSTEP));
      uint4*       dst = (uint4*)&Bs[0][0];
      for (int idx = tid; idx < (DIM * KSTEP) / 8; idx += 256) dst[idx] = src[idx];
    }
    if (ks + 1 < NKS)
      __builtin_prefetch(wtp + (size_t)(ks + 1) * (DIM * KSTEP), 0, 1);
    __syncthreads();

    v16bf afrag[2];
#pragma unroll
    for (int mi = 0; mi < 2; ++mi)
      afrag[mi] = load_a_frag(&As[(waveM * 2 + mi) * 16 + lm][0], hf);
#pragma unroll
    for (int ni = 0; ni < 6; ++ni) {
      v16bf bfrag = load_b_frag(&Bs[(waveN * 6 + ni) * 16 + lm][0], hf);
#pragma unroll
      for (int mi = 0; mi < 2; ++mi)
        acc[mi][ni] = __builtin_amdgcn_wmma_f32_16x16x32_bf16(
            false, afrag[mi], false, bfrag, (short)0, acc[mi][ni], false, false);
    }
    __syncthreads();
  }

  const float scale = (p == 0) ? 0.17677669529663687f : 1.0f;  // hd^-0.5 for Q
#pragma unroll
  for (int mi = 0; mi < 2; ++mi)
#pragma unroll
    for (int ni = 0; ni < 6; ++ni) {
      int n = (waveN * 6 + ni) * 16 + lm;
      int head = n >> 5, d = n & 31;
#pragma unroll
      for (int r = 0; r < 8; ++r) {
        int m = (waveM * 2 + mi) * 16 + hf * 8 + r;
        float v = (acc[mi][ni][r] + bias[n]) * scale;
        if (p == 2)  // V transposed: [win][head][hd][tok]
          outp[(((size_t)win * NHEAD + head) * HD + d) * NTOK + m] = (bf16)v;
        else         // Q/K: [win][head][tok][hd]
          outp[(((size_t)win * NHEAD + head) * NTOK + m) * HD + d] = (bf16)v;
      }
    }
}

// ---------------------------------------------------------------------------
// Kernel 2: attention, one block (4 waves / 128 thr) per (window, head).
// S = Q K^T (16 WMMAs) -> +rel_bias +shift_mask -> softmax -> P V (16 WMMAs).
// Relative-position bias column staged to LDS; shift mask computed
// analytically (region ids), no mask tensor.
// ---------------------------------------------------------------------------
__global__ void __launch_bounds__(128)
k_attn(const bf16* __restrict__ q_ws, const bf16* __restrict__ k_ws,
       const bf16* __restrict__ v_ws, const float* __restrict__ bias_table,
       bf16* __restrict__ attn_ws) {
  __shared__ __align__(16) bf16 Qs[64][32];
  __shared__ __align__(16) bf16 Ks[64][32];
  __shared__ __align__(16) bf16 Vt[32][64];     // V transposed: [d][token]
  __shared__ float Sf[64][64];
  __shared__ __align__(16) bf16 Ps[64][64];
  __shared__ float bt[(2 * WSZ - 1) * (2 * WSZ - 1)];  // this head's bias column

  const int win = blockIdx.x;
  const int h   = blockIdx.y;
  const int wy  = (win & 63) >> 3;
  const int wx  = win & 7;
  const int tid = threadIdx.x;
  const int wave = tid >> 5, lane = tid & 31;
  const int hf = lane >> 4, lm = lane & 15;

  // stage Q/K/Vt tiles (64x32 bf16 = 4KB each) as 128-bit copies
  {
    size_t base = ((size_t)win * NHEAD + h) * (NTOK * HD);
    const uint4* qs = (const uint4*)(q_ws + base);
    const uint4* ks = (const uint4*)(k_ws + base);
    const uint4* vs = (const uint4*)(v_ws + base);  // already transposed
    uint4* qd = (uint4*)&Qs[0][0];
    uint4* kd = (uint4*)&Ks[0][0];
    uint4* vd = (uint4*)&Vt[0][0];
    for (int i = tid; i < 256; i += 128) { qd[i] = qs[i]; kd[i] = ks[i]; vd[i] = vs[i]; }
    for (int i = tid; i < (2 * WSZ - 1) * (2 * WSZ - 1); i += 128)
      bt[i] = bias_table[i * NHEAD + h];
  }
  __syncthreads();

  // ---- S = Q @ K^T : wave owns M-tile `wave`, all 4 N-tiles, K=32 ----
  v16bf qa = load_a_frag(&Qs[wave * 16 + lm][0], hf);
  v8f zero = {};
#pragma unroll
  for (int nt = 0; nt < 4; ++nt) {
    // B[k=d][n=token j] = K[j][d]: K-contiguous run of row j of Ks
    v16bf kb = load_b_frag(&Ks[nt * 16 + lm][0], hf);
    v8f sc = __builtin_amdgcn_wmma_f32_16x16x32_bf16(
        false, qa, false, kb, (short)0, zero, false, false);
#pragma unroll
    for (int r = 0; r < 8; ++r) Sf[wave * 16 + hf * 8 + r][nt * 16 + lm] = sc[r];
  }
  __syncthreads();

  // ---- softmax (one row per thread, rows 0..63) with bias + shift mask ----
  if (tid < 64) {
    int i  = tid;
    int yi = i >> 3, xi = i & 7;
    int id_i = region_id(wy * WSZ + yi) * 3 + region_id(wx * WSZ + xi);
    float mx = -1e30f;
    for (int j = 0; j < 64; ++j) {
      int yj = j >> 3, xj = j & 7;
      int ridx = (yi - yj + WSZ - 1) * (2 * WSZ - 1) + (xi - xj + WSZ - 1);
      float s = Sf[i][j] + bt[ridx];
      int id_j = region_id(wy * WSZ + yj) * 3 + region_id(wx * WSZ + xj);
      if (id_j != id_i) s -= 100.0f;
      Sf[i][j] = s;
      mx = fmaxf(mx, s);
    }
    float sum = 0.0f;
    for (int j = 0; j < 64; ++j) { float e = __expf(Sf[i][j] - mx); Sf[i][j] = e; sum += e; }
    float rinv = 1.0f / sum;
    for (int j = 0; j < 64; ++j) Ps[i][j] = (bf16)(Sf[i][j] * rinv);
  }
  __syncthreads();

  // ---- O = P @ V : wave owns M-tile `wave`, 2 N-tiles (hd=32), 2 K-steps ----
  v8f oc[2] = {zero, zero};
#pragma unroll
  for (int ks = 0; ks < 2; ++ks) {
    v16bf pa = load_a_frag(&Ps[wave * 16 + lm][ks * 32], hf);
#pragma unroll
    for (int nt = 0; nt < 2; ++nt) {
      // B[k=token][n=d] = V[token][d] = Vt[d][token]: K-contiguous in Vt row
      v16bf vb = load_b_frag(&Vt[nt * 16 + lm][ks * 32], hf);
      oc[nt] = __builtin_amdgcn_wmma_f32_16x16x32_bf16(
          false, pa, false, vb, (short)0, oc[nt], false, false);
    }
  }
#pragma unroll
  for (int nt = 0; nt < 2; ++nt)
#pragma unroll
    for (int r = 0; r < 8; ++r) {
      int m = wave * 16 + hf * 8 + r;
      int d = nt * 16 + lm;
      attn_ws[((size_t)win * NTOK + m) * DIM + h * HD + d] = (bf16)oc[nt][r];
    }
}

// ---------------------------------------------------------------------------
// Kernel 3: output projection + window reverse + roll(+4,+4), f32 out.
// Same tiling as k_qkv; A is already bf16 in the workspace.
// ---------------------------------------------------------------------------
__global__ void __launch_bounds__(256)
k_oproj(const bf16* __restrict__ attn_ws, const bf16* __restrict__ wt,
        const float* __restrict__ bo, float* __restrict__ out) {
  __shared__ __align__(16) bf16 As[64][KSTEP];
  __shared__ __align__(16) bf16 Bs[DIM][KSTEP];

  const int win = blockIdx.x;
  const int b   = win >> 6;
  const int wy  = (win & 63) >> 3;
  const int wx  = win & 7;
  const int tid = threadIdx.x;
  const int wave = tid >> 5, lane = tid & 31;
  const int hf = lane >> 4, lm = lane & 15;
  const int waveM = wave >> 2;
  const int waveN = wave & 3;

  const bf16* wtp   = wt + (size_t)3 * WMAT;   // wo, packed K-step-major
  const bf16* abase = attn_ws + (size_t)win * NTOK * DIM;

  v8f zero = {};
  v8f acc[2][6];
#pragma unroll
  for (int i = 0; i < 2; ++i)
#pragma unroll
    for (int j = 0; j < 6; ++j) acc[i][j] = zero;

  for (int ks = 0; ks < NKS; ++ks) {
    const int k0 = ks * KSTEP;
    // A tile: 64 rows x 32 cols bf16 (row stride DIM), 128-bit loads
    for (int idx = tid; idx < 256; idx += 256) {
      int m = idx >> 2, q4 = idx & 3;
      ((uint4*)&As[m][0])[q4] =
          *((const uint4*)(abase + (size_t)m * DIM + k0) + q4);
    }
    {
      const uint4* src = (const uint4*)(wtp + (size_t)ks * (DIM * KSTEP));
      uint4*       dst = (uint4*)&Bs[0][0];
      for (int idx = tid; idx < (DIM * KSTEP) / 8; idx += 256) dst[idx] = src[idx];
    }
    if (ks + 1 < NKS)
      __builtin_prefetch(wtp + (size_t)(ks + 1) * (DIM * KSTEP), 0, 1);
    __syncthreads();

    v16bf afrag[2];
#pragma unroll
    for (int mi = 0; mi < 2; ++mi)
      afrag[mi] = load_a_frag(&As[(waveM * 2 + mi) * 16 + lm][0], hf);
#pragma unroll
    for (int ni = 0; ni < 6; ++ni) {
      v16bf bfrag = load_b_frag(&Bs[(waveN * 6 + ni) * 16 + lm][0], hf);
#pragma unroll
      for (int mi = 0; mi < 2; ++mi)
        acc[mi][ni] = __builtin_amdgcn_wmma_f32_16x16x32_bf16(
            false, afrag[mi], false, bfrag, (short)0, acc[mi][ni], false, false);
    }
    __syncthreads();
  }

#pragma unroll
  for (int mi = 0; mi < 2; ++mi)
#pragma unroll
    for (int ni = 0; ni < 6; ++ni) {
      int n = (waveN * 6 + ni) * 16 + lm;
#pragma unroll
      for (int r = 0; r < 8; ++r) {
        int m = (waveM * 2 + mi) * 16 + hf * 8 + r;
        int hs = wy * WSZ + (m >> 3), vs = wx * WSZ + (m & 7);
        int hd = (hs + SHIFT) & (IMG - 1);     // roll(+4,+4) on the way out
        int wd = (vs + SHIFT) & (IMG - 1);
        out[(((size_t)b * IMG + hd) * IMG + wd) * DIM + n] = acc[mi][ni][r] + bo[n];
      }
    }
}

// ---------------------------------------------------------------------------
extern "C" void kernel_launch(void* const* d_in, const int* in_sizes, int n_in,
                              void* d_out, int out_size, void* d_ws, size_t ws_size,
                              hipStream_t stream) {
  const float* x  = (const float*)d_in[0];
  const float* wq = (const float*)d_in[1];
  const float* bq = (const float*)d_in[2];
  const float* wk = (const float*)d_in[3];
  const float* bk = (const float*)d_in[4];
  const float* wv = (const float*)d_in[5];
  const float* bv = (const float*)d_in[6];
  const float* wo = (const float*)d_in[7];
  const float* bo = (const float*)d_in[8];
  const float* bias_table = (const float*)d_in[9];

  // Workspace layout (bytes):
  //   wt:      4 * 147456 bf16               =   1,179,648
  //   q/k/v:   3 * 2048*12*64*32 bf16        = 3*100,663,296
  //   attn:    2048*64*384 bf16              =   100,663,296
  // total ~404 MB.
  char* ws = (char*)d_ws;
  size_t off = 0;
  bf16* wt = (bf16*)(ws + off);
  off += (size_t)4 * WMAT * sizeof(bf16);
  off = (off + 255) & ~(size_t)255;
  const size_t qkv_bytes = (size_t)NWIN * NHEAD * NTOK * HD * sizeof(bf16);
  bf16* q_ws = (bf16*)(ws + off); off += qkv_bytes;
  bf16* k_ws = (bf16*)(ws + off); off += qkv_bytes;
  bf16* v_ws = (bf16*)(ws + off); off += qkv_bytes;
  bf16* attn_ws = (bf16*)(ws + off);

  k_pack_weights<<<(4 * WMAT + 255) / 256, 256, 0, stream>>>(wq, wk, wv, wo, wt);
  k_qkv<<<dim3(NWIN, 3), 256, 0, stream>>>(x, wt, bq, bk, bv, q_ws, k_ws, v_ws);
  k_attn<<<dim3(NWIN, NHEAD), 128, 0, stream>>>(q_ws, k_ws, v_ws, bias_table, attn_ws);
  k_oproj<<<NWIN, 256, 0, stream>>>(attn_ws, wt, bo, (float*)d_out);
}
